// UserBehaviorGNN_53068615909745
// MI455X (gfx1250) — matrix operand order repaired and checked by reading, not compiled, not run on hardware
//
#include <hip/hip_runtime.h>
#include <hip/hip_bf16.h>

typedef __attribute__((ext_vector_type(16))) _Float16 v16h;
typedef __attribute__((ext_vector_type(8)))  _Float16 v8h;
typedef __attribute__((ext_vector_type(8)))  float    v8f;
typedef __attribute__((ext_vector_type(4)))  float    v4f;

// ---------------------------------------------------------------------------
// Weight packing: W is [K x 64] row-major f32. Pack into per-fragment f16
// blocks matching the CDNA5 WMMA 16-bit B-matrix (KxN = 32x16) lane layout:
//   fragment f = kstep*4 + ntile  (each fragment = 32 lanes x 16 halves)
//   lane L: column n = ntile*16 + (L&15); K rows = kstep*32 + (j>>3)*16 + (L>>4)*8 + (j&7)
// Hot kernels then fetch a lane's fragment with two global_load_b128.
// ---------------------------------------------------------------------------
__global__ void pack_w_kernel(const float* __restrict__ W,
                              _Float16* __restrict__ out, int K) {
    int tid = blockIdx.x * blockDim.x + threadIdx.x;
    int total = K * 64;
    if (tid >= total) return;
    int f    = tid >> 9;        // fragment index (512 halves each)
    int rem  = tid & 511;
    int lane = rem >> 4;
    int j    = rem & 15;
    int kstep = f >> 2;
    int ntile = f & 3;
    int n = ntile * 16 + (lane & 15);
    int k = kstep * 32 + (j >> 3) * 16 + (lane >> 4) * 8 + (j & 7);
    out[tid] = (_Float16)W[(size_t)k * 64 + n];
}

// Histogram of edge targets (fp32 counts so the mean divide is a simple rcp)
__global__ void count_kernel(const int* __restrict__ idx,
                             float* __restrict__ cnt, int n) {
    int i = blockIdx.x * blockDim.x + threadIdx.x;
    int stride = gridDim.x * blockDim.x;
    for (; i < n; i += stride) atomicAdd(&cnt[idx[i]], 1.0f);
}

// ---------------------------------------------------------------------------
// Embed kernel: out[r,:] (or scatter acc[idx[r],:]) = relu(x[r,:] @ W + b)
// One wave per 16-row tile; K_IN/32 k-steps x 4 n-tiles of
// v_wmma_f32_16x16x32_f16. A-fragment per ISA 16-bit layout:
//   lane half h = L>>4, row r = L&15:
//   elements 0..7 = K k0+h*8..+7, elements 8..15 = K k0+16+h*8..+7
// ---------------------------------------------------------------------------
template <int K_IN, bool SCATTER>
__global__ void __launch_bounds__(128)
embed_kernel(const float* __restrict__ x, const _Float16* __restrict__ pW,
             const float* __restrict__ bias, const int* __restrict__ idx,
             float* __restrict__ out, int n_rows) {
    const int lane = threadIdx.x & 31;
    const int wave = threadIdx.x >> 5;
    const int rowbase = (blockIdx.x * 4 + wave) * 16;
    if (rowbase >= n_rows) return;   // uniform per wave: EXEC stays all-1s
    const int r = lane & 15;
    const int h = lane >> 4;
    const float* xrow = x + (size_t)(rowbase + r) * K_IN;

    v8f c[4] = {v8f{0}, v8f{0}, v8f{0}, v8f{0}};

#pragma unroll
    for (int ks = 0; ks < K_IN / 32; ++ks) {
        const int k0 = ks * 32;
        const v4f x0 = *(const v4f*)(xrow + k0 + h * 8);
        const v4f x1 = *(const v4f*)(xrow + k0 + h * 8 + 4);
        const v4f x2 = *(const v4f*)(xrow + k0 + 16 + h * 8);
        const v4f x3 = *(const v4f*)(xrow + k0 + 16 + h * 8 + 4);
        v16h a;
#pragma unroll
        for (int j = 0; j < 4; ++j) {
            a[j]      = (_Float16)x0[j];
            a[j + 4]  = (_Float16)x1[j];
            a[j + 8]  = (_Float16)x2[j];
            a[j + 12] = (_Float16)x3[j];
        }
#pragma unroll
        for (int n = 0; n < 4; ++n) {
            const v16h b = *(const v16h*)(pW + (((size_t)(ks * 4 + n)) << 9) + lane * 16);
            c[n] = __builtin_amdgcn_wmma_f32_16x16x32_f16(
                false, a, false, b, (short)0, c[n], false, false);
        }
    }

    int tg[8];
    if (SCATTER) {
#pragma unroll
        for (int e = 0; e < 8; ++e) tg[e] = idx[rowbase + h * 8 + e];
    }
    // C/D layout: lane holds column (L&15) of n-tile, rows (L>>4)*8 + e
#pragma unroll
    for (int n = 0; n < 4; ++n) {
        const float bv = bias[n * 16 + r];
#pragma unroll
        for (int e = 0; e < 8; ++e) {
            float v = c[n][e] + bv;
            v = v > 0.f ? v : 0.f;
            if (SCATTER) {
                atomicAdd(out + (size_t)tg[e] * 64 + n * 16 + r, v);
            } else {
                out[(size_t)(rowbase + h * 8 + e) * 64 + n * 16 + r] = v;
            }
        }
    }
}

// ---------------------------------------------------------------------------
// Fused MLP: fused(256) = [user_h | mean_s | mean_m | mean_f]
//   h   = relu(fused @ Wf1 + bf1)   (WMMA, K=256)
//   emb = relu(h @ Wf2 + bf2)       (WMMA, K=64, via wave-private LDS relayout)
//   logit = emb @ Wc + bc           (lanes 0..15, 64-wide dot from LDS)
// ---------------------------------------------------------------------------
__global__ void __launch_bounds__(128)
fuse_kernel(const float* __restrict__ user_h,
            const float* __restrict__ acc_s, const float* __restrict__ cnt_s,
            const float* __restrict__ acc_m, const float* __restrict__ cnt_m,
            const float* __restrict__ acc_f, const float* __restrict__ cnt_f,
            const _Float16* __restrict__ pWf1, const float* __restrict__ bf1,
            const _Float16* __restrict__ pWf2, const float* __restrict__ bf2,
            const float* __restrict__ Wc, const float* __restrict__ bc,
            float* __restrict__ logits, float* __restrict__ emb, int n_users) {
    __shared__ _Float16 lds_h[4][16 * 64];
    __shared__ float    lds_e[4][16 * 64];

    const int lane = threadIdx.x & 31;
    const int wave = threadIdx.x >> 5;
    const int rowbase = (blockIdx.x * 4 + wave) * 16;
    if (rowbase >= n_users) return;
    const int r = lane & 15;
    const int h = lane >> 4;

    const float* srcs[4] = {user_h, acc_s, acc_m, acc_f};
    float scl[4];
    scl[0] = 1.f;
    scl[1] = 1.f / fmaxf(cnt_s[rowbase + r], 1.f);
    scl[2] = 1.f / fmaxf(cnt_m[rowbase + r], 1.f);
    scl[3] = 1.f / fmaxf(cnt_f[rowbase + r], 1.f);

    // ---- GEMM1: fused[16,256] @ Wf1[256,64] ----
    v8f c[4] = {v8f{0}, v8f{0}, v8f{0}, v8f{0}};
#pragma unroll
    for (int ks = 0; ks < 8; ++ks) {
        const int blk = ks >> 1;
        const int k0  = (ks & 1) * 32;
        const float* srow = srcs[blk] + (size_t)(rowbase + r) * 64 + k0;
        const float s = scl[blk];
        const v4f x0 = *(const v4f*)(srow + h * 8);
        const v4f x1 = *(const v4f*)(srow + h * 8 + 4);
        const v4f x2 = *(const v4f*)(srow + 16 + h * 8);
        const v4f x3 = *(const v4f*)(srow + 16 + h * 8 + 4);
        v16h a;
#pragma unroll
        for (int j = 0; j < 4; ++j) {
            a[j]      = (_Float16)(x0[j] * s);
            a[j + 4]  = (_Float16)(x1[j] * s);
            a[j + 8]  = (_Float16)(x2[j] * s);
            a[j + 12] = (_Float16)(x3[j] * s);
        }
#pragma unroll
        for (int n = 0; n < 4; ++n) {
            const v16h b = *(const v16h*)(pWf1 + (((size_t)(ks * 4 + n)) << 9) + lane * 16);
            c[n] = __builtin_amdgcn_wmma_f32_16x16x32_f16(
                false, a, false, b, (short)0, c[n], false, false);
        }
    }
    // bias + relu, re-layout C->A through wave-private LDS (f16)
#pragma unroll
    for (int n = 0; n < 4; ++n) {
        const float bv = bf1[n * 16 + r];
#pragma unroll
        for (int e = 0; e < 8; ++e) {
            float v = c[n][e] + bv;
            v = v > 0.f ? v : 0.f;
            lds_h[wave][(h * 8 + e) * 64 + n * 16 + r] = (_Float16)v;
        }
    }
    asm volatile("s_wait_dscnt 0" ::: "memory");  // same-wave LDS RAW

    // ---- GEMM2: h[16,64] @ Wf2[64,64] ----
    v8f c2[4] = {v8f{0}, v8f{0}, v8f{0}, v8f{0}};
#pragma unroll
    for (int ks = 0; ks < 2; ++ks) {
        const int k0 = ks * 32;
        const v8h lo = *(const v8h*)&lds_h[wave][r * 64 + k0 + h * 8];
        const v8h hi = *(const v8h*)&lds_h[wave][r * 64 + k0 + 16 + h * 8];
        const v16h a = __builtin_shufflevector(lo, hi, 0, 1, 2, 3, 4, 5, 6, 7,
                                               8, 9, 10, 11, 12, 13, 14, 15);
#pragma unroll
        for (int n = 0; n < 4; ++n) {
            const v16h b = *(const v16h*)(pWf2 + (((size_t)(ks * 4 + n)) << 9) + lane * 16);
            c2[n] = __builtin_amdgcn_wmma_f32_16x16x32_f16(
                false, a, false, b, (short)0, c2[n], false, false);
        }
    }
#pragma unroll
    for (int n = 0; n < 4; ++n) {
        const float bv = bf2[n * 16 + r];
#pragma unroll
        for (int e = 0; e < 8; ++e) {
            float v = c2[n][e] + bv;
            v = v > 0.f ? v : 0.f;
            const int m = h * 8 + e, col = n * 16 + r;
            emb[(size_t)(rowbase + m) * 64 + col] = v;
            lds_e[wave][m * 64 + col] = v;
        }
    }
    asm volatile("s_wait_dscnt 0" ::: "memory");

    // ---- logits: emb[16,64] @ Wc[64,1] ----
    if (lane < 16) {
        float s = bc[0];
#pragma unroll 8
        for (int j = 0; j < 64; ++j) s += lds_e[wave][lane * 64 + j] * Wc[j];
        logits[rowbase + lane] = s;
    }
}

// ---------------------------------------------------------------------------
extern "C" void kernel_launch(void* const* d_in, const int* in_sizes, int n_in,
                              void* d_out, int out_size, void* d_ws, size_t ws_size,
                              hipStream_t stream) {
    const float* user_x     = (const float*)d_in[0];
    const float* session_x  = (const float*)d_in[1];
    const float* message_x  = (const float*)d_in[2];
    const float* feedback_x = (const float*)d_in[3];
    const int*   s2u        = (const int*)d_in[4];
    const int*   m2u        = (const int*)d_in[5];
    const int*   f2u        = (const int*)d_in[6];
    const float* W_user = (const float*)d_in[7],  *b_user = (const float*)d_in[8];
    const float* W_sess = (const float*)d_in[9],  *b_sess = (const float*)d_in[10];
    const float* W_msg  = (const float*)d_in[11], *b_msg  = (const float*)d_in[12];
    const float* W_fb   = (const float*)d_in[13], *b_fb   = (const float*)d_in[14];
    const float* W_f1   = (const float*)d_in[15], *b_f1   = (const float*)d_in[16];
    const float* W_f2   = (const float*)d_in[17], *b_f2   = (const float*)d_in[18];
    const float* W_c    = (const float*)d_in[19], *b_c    = (const float*)d_in[20];

    const int n_users = in_sizes[0] / 128;
    const int n_sess  = in_sizes[1] / 64;
    const int n_msg   = in_sizes[2] / 64;
    const int n_fb    = in_sizes[3] / 32;

    // ---- workspace carve (accs+counts contiguous for a single memset) ----
    char* ws = (char*)d_ws;
    size_t off = 0;
    auto carve = [&](size_t bytes) -> void* {
        void* p = ws + off;
        off = (off + bytes + 255) & ~(size_t)255;
        return p;
    };
    const size_t accB = (size_t)n_users * 64 * sizeof(float);
    const size_t cntB = (size_t)n_users * sizeof(float);
    float* acc_s = (float*)carve(accB);
    float* acc_m = (float*)carve(accB);
    float* acc_f = (float*)carve(accB);
    float* cnt_s = (float*)carve(cntB);
    float* cnt_m = (float*)carve(cntB);
    float* cnt_f = (float*)carve(cntB);
    const size_t zero_bytes = (size_t)((char*)(cnt_f) - (char*)acc_s) + cntB;
    float* uh = (float*)carve(accB);
    _Float16* pWu  = (_Float16*)carve((size_t)128 * 64 * 2);
    _Float16* pWs  = (_Float16*)carve((size_t)64 * 64 * 2);
    _Float16* pWm  = (_Float16*)carve((size_t)64 * 64 * 2);
    _Float16* pWfb = (_Float16*)carve((size_t)32 * 64 * 2);
    _Float16* pWf1 = (_Float16*)carve((size_t)256 * 64 * 2);
    _Float16* pWf2 = (_Float16*)carve((size_t)64 * 64 * 2);

    hipMemsetAsync(acc_s, 0, zero_bytes, stream);

    // ---- pack weights into WMMA B-fragment layout ----
    auto packGrid = [](int K) { return dim3((K * 64 + 255) / 256); };
    pack_w_kernel<<<packGrid(128), 256, 0, stream>>>(W_user, pWu, 128);
    pack_w_kernel<<<packGrid(64),  256, 0, stream>>>(W_sess, pWs, 64);
    pack_w_kernel<<<packGrid(64),  256, 0, stream>>>(W_msg,  pWm, 64);
    pack_w_kernel<<<packGrid(32),  256, 0, stream>>>(W_fb,   pWfb, 32);
    pack_w_kernel<<<packGrid(256), 256, 0, stream>>>(W_f1,   pWf1, 256);
    pack_w_kernel<<<packGrid(64),  256, 0, stream>>>(W_f2,   pWf2, 64);

    // ---- edge-count histograms (overlap with embeds in issue order) ----
    count_kernel<<<2048, 256, 0, stream>>>(s2u, cnt_s, n_sess);
    count_kernel<<<2048, 256, 0, stream>>>(m2u, cnt_m, n_msg);
    count_kernel<<<2048, 256, 0, stream>>>(f2u, cnt_f, n_fb);

    // ---- per-node embeds: 4 waves/block, one 16-row tile per wave ----
    auto gridFor = [](int rows) { return dim3((rows / 16 + 3) / 4); };
    embed_kernel<128, false><<<gridFor(n_users), 128, 0, stream>>>(
        user_x, pWu, b_user, nullptr, uh, n_users);
    embed_kernel<64, true><<<gridFor(n_sess), 128, 0, stream>>>(
        session_x, pWs, b_sess, s2u, acc_s, n_sess);
    embed_kernel<64, true><<<gridFor(n_msg), 128, 0, stream>>>(
        message_x, pWm, b_msg, m2u, acc_m, n_msg);
    embed_kernel<32, true><<<gridFor(n_fb), 128, 0, stream>>>(
        feedback_x, pWfb, b_fb, f2u, acc_f, n_fb);

    // ---- fused MLP + classifier ----
    float* logits = (float*)d_out;
    float* emb    = logits + n_users;
    fuse_kernel<<<gridFor(n_users), 128, 0, stream>>>(
        uh, acc_s, cnt_s, acc_m, cnt_m, acc_f, cnt_f,
        pWf1, b_f1, pWf2, b_f2, W_c, b_c, logits, emb, n_users);
}
